// RENBase_64656437674040
// MI455X (gfx1250) — matrix operand order repaired and checked by reading, not compiled, not run on hardware
//
#include <hip/hip_runtime.h>
#include <hip/hip_bf16.h>

// REN forward for MI455X (gfx1250): bf16 split-precision WMMA GEMMs
// (v_wmma_f32_16x16x32_bf16) with double-buffered async global->LDS copies
// (global_load_async_to_lds_b128 + s_wait_asynccnt), blocked triangular
// equilibrium, 1-WG LU.

#define NXc   512
#define NVc   1024
#define NUc   256
#define NYc   256
#define NT    2048
#define BAT   4096
#define EPSC  1.1920929e-07f

typedef __bf16 bf16_t;
typedef __attribute__((ext_vector_type(16))) __bf16 v16bf;
typedef __attribute__((ext_vector_type(8)))  float  v8f;

__device__ __forceinline__ void splitf(float x, bf16_t& h, bf16_t& l) {
  h = (bf16_t)x;
  l = (bf16_t)(x - (float)h);
}

// Async 16B copy global -> LDS (CDNA5): tracked by ASYNCcnt.
// LDS dst operand is the low 32 bits of the flat address (LDS aperture offset).
__device__ __forceinline__ void async_ld_b128(const bf16_t* g, bf16_t* lds) {
  asm volatile("global_load_async_to_lds_b128 %0, %1, off"
               :: "v"((unsigned)(uintptr_t)lds),
                  "v"((unsigned long long)(uintptr_t)g)
               : "memory");
}

// ---------------------------------------------------------------------------
// Generic NT GEMM:  Cout[m][n] = scale * sum_k A[m][k]*B[n][k]  (+ Cin + bias
// + eps*diag).  A,B given as bf16 hi/lo splits; product uses 3 WMMA terms:
// Ahi*Bhi + Ahi*Blo + Alo*Bhi  (~fp32 accuracy at bf16 WMMA rate).
// Grid: (N/64, M/64), block 128 threads = 4 waves; wave -> 32x32 of C.
// All M,N multiples of 64; K multiple of 32 (K==0 legal -> pure epilogue).
// Tiles are double-buffered in LDS and filled with async copies.
// ---------------------------------------------------------------------------
__global__ __launch_bounds__(128) void ren_gemm_nt(
    const bf16_t* __restrict__ Ah, const bf16_t* __restrict__ Al, int lda,
    const bf16_t* __restrict__ Bh, const bf16_t* __restrict__ Bl, int ldb,
    const float* __restrict__ Cin, int ldcin,
    float* __restrict__ Cout, int ldcout,
    const float* __restrict__ bias,
    const float* __restrict__ pscale, const float* __restrict__ nrmp,
    int K, int addDiagEps)
{
  __shared__ __align__(16) bf16_t sA[2][2][64][32]; // [buf][hi/lo][row][k]
  __shared__ __align__(16) bf16_t sB[2][2][64][32];

  const int tid  = threadIdx.x;
  const int lane = tid & 31;
  const int wave = tid >> 5;
  const int wr   = (wave >> 1) * 32;
  const int wc   = (wave & 1) * 32;
  const int m0   = blockIdx.y * 64;
  const int n0   = blockIdx.x * 64;

  // async staging role for this thread: row r (0..63), 32B half c0 of the row
  const int r  = tid >> 1;
  const int c0 = (tid & 1) * 16;     // in bf16 elements

  v8f acc[2][2] = {};
  const int ksteps = K >> 5;

  auto issue = [&](int buf, int kt) {
    const int k0 = kt << 5;
    const bf16_t* ga0 = Ah + (size_t)(m0 + r) * lda + k0 + c0;
    const bf16_t* ga1 = Al + (size_t)(m0 + r) * lda + k0 + c0;
    const bf16_t* gb0 = Bh + (size_t)(n0 + r) * ldb + k0 + c0;
    const bf16_t* gb1 = Bl + (size_t)(n0 + r) * ldb + k0 + c0;
    async_ld_b128(ga0,     &sA[buf][0][r][c0]);
    async_ld_b128(ga0 + 8, &sA[buf][0][r][c0 + 8]);
    async_ld_b128(ga1,     &sA[buf][1][r][c0]);
    async_ld_b128(ga1 + 8, &sA[buf][1][r][c0 + 8]);
    async_ld_b128(gb0,     &sB[buf][0][r][c0]);
    async_ld_b128(gb0 + 8, &sB[buf][0][r][c0 + 8]);
    async_ld_b128(gb1,     &sB[buf][1][r][c0]);
    async_ld_b128(gb1 + 8, &sB[buf][1][r][c0 + 8]);
  };

  if (ksteps > 0) issue(0, 0);

  for (int kt = 0; kt < ksteps; ++kt) {
    const int b = kt & 1;
    if (kt + 1 < ksteps) {
      issue(1 - b, kt + 1);                       // prefetch next K-step
      asm volatile("s_wait_asynccnt 0x8" ::: "memory");  // older batch done
    } else {
      asm volatile("s_wait_asynccnt 0x0" ::: "memory");
    }
    __syncthreads();

    // Build WMMA fragments per ISA 16-bit A (16x32) / B (32x16) lane layouts.
    v16bf ah[2], al[2], bh[2], bl[2];
    {
      const int mr  = lane & 15;
      const int akb = (lane >= 16) ? 8  : 0;
      const int bkb = (lane >= 16) ? 16 : 0;
      for (int s = 0; s < 2; ++s) {
        union { v16bf v; unsigned u[8]; } fh, fl;
        const int m = wr + s * 16 + mr;
        for (int vv = 0; vv < 8; ++vv) {
          const int kk = ((vv < 4) ? 0 : 16) + akb + 2 * (vv & 3);
          fh.u[vv] = *(const unsigned*)&sA[b][0][m][kk];
          fl.u[vv] = *(const unsigned*)&sA[b][1][m][kk];
        }
        ah[s] = fh.v; al[s] = fl.v;
        const int n = wc + s * 16 + mr;
        for (int vv = 0; vv < 8; ++vv) {
          const int kk = ((vv < 4) ? 0 : 16) + (bkb >> 1) + 2 * (vv & 3);
          fh.u[vv] = *(const unsigned*)&sB[b][0][n][kk];
          fl.u[vv] = *(const unsigned*)&sB[b][1][n][kk];
        }
        bh[s] = fh.v; bl[s] = fl.v;
      }
    }
    for (int mi = 0; mi < 2; ++mi)
      for (int ni = 0; ni < 2; ++ni) {
        acc[mi][ni] = __builtin_amdgcn_wmma_f32_16x16x32_bf16(
            false, ah[mi], false, bh[ni], (short)0, acc[mi][ni], false, false);
        acc[mi][ni] = __builtin_amdgcn_wmma_f32_16x16x32_bf16(
            false, al[mi], false, bh[ni], (short)0, acc[mi][ni], false, false);
        acc[mi][ni] = __builtin_amdgcn_wmma_f32_16x16x32_bf16(
            false, ah[mi], false, bl[ni], (short)0, acc[mi][ni], false, false);
      }
    __syncthreads();   // all waves done reading buf b before it is refilled
  }

  float scale = 1.0f;
  if (pscale) { float n2 = fmaxf(nrmp[0], EPSC); scale = pscale[0] * pscale[0] / n2; }

  for (int mi = 0; mi < 2; ++mi)
    for (int ni = 0; ni < 2; ++ni)
      for (int rr = 0; rr < 8; ++rr) {
        int gm = m0 + wr + mi * 16 + rr + ((lane >= 16) ? 8 : 0);
        int gn = n0 + wc + ni * 16 + (lane & 15);
        float v = scale * acc[mi][ni][rr];
        if (Cin)  v += Cin[(size_t)gm * ldcin + gn];
        if (bias) v += bias[gn];
        if (addDiagEps && gm == gn) v += EPSC;
        Cout[(size_t)gm * ldcout + gn] = v;
      }
}

// ---------------------------------------------------------------------------
__global__ void ren_init(float* nrm) { if (threadIdx.x == 0) nrm[0] = 0.0f; }

__global__ __launch_bounds__(256) void ren_sumsq(const float* __restrict__ X, float* nrm) {
  __shared__ float red[256];
  size_t i      = (size_t)blockIdx.x * blockDim.x + threadIdx.x;
  size_t stride = (size_t)gridDim.x * blockDim.x;
  float s = 0.0f;
  for (; i < (size_t)NT * NT; i += stride) { float v = X[i]; s += v * v; }
  red[threadIdx.x] = s; __syncthreads();
  for (int w = 128; w > 0; w >>= 1) {
    if (threadIdx.x < w) red[threadIdx.x] += red[threadIdx.x + w];
    __syncthreads();
  }
  if (threadIdx.x == 0) atomicAdd(nrm, red[0]);
}

// Xt[i][k] = X[k][i], stored as bf16 hi/lo (both GEMM operands of X^T X).
__global__ void ren_split_tr(const float* __restrict__ X,
                             bf16_t* __restrict__ th, bf16_t* __restrict__ tl) {
  __shared__ float t[16][17];
  int bx = blockIdx.x * 16, by = blockIdx.y * 16;
  int tx = threadIdx.x,     ty = threadIdx.y;
  t[ty][tx] = X[(size_t)(bx + ty) * NT + (by + tx)];
  __syncthreads();
  float v = t[tx][ty];                      // = X[bx+tx][by+ty]
  bf16_t h, l; splitf(v, h, l);
  size_t o = (size_t)(by + ty) * NT + (bx + tx);
  th[o] = h; tl[o] = l;
}

__global__ void ren_split(const float* __restrict__ src,
                          bf16_t* __restrict__ hi, bf16_t* __restrict__ lo, int n) {
  int i = blockIdx.x * 256 + threadIdx.x;
  if (i < n) { bf16_t h, l; splitf(src[i], h, l); hi[i] = h; lo[i] = l; }
}

// E = 0.5*(H11 + H33 + Y1 - Y1^T)
__global__ void ren_buildE(const float* __restrict__ H, const float* __restrict__ Y1,
                           float* __restrict__ E) {
  int j = blockIdx.x * 16 + threadIdx.x;
  int i = blockIdx.y * 16 + threadIdx.y;
  E[(size_t)i * NXc + j] = 0.5f * (H[(size_t)i * NT + j]
                                 + H[(size_t)(NXc + NVc + i) * NT + (NXc + NVc + j)]
                                 + Y1[(size_t)i * NXc + j]
                                 - Y1[(size_t)j * NXc + i]);
}

// In-place right-looking LU (no pivoting; sym-part of E is PD). One workgroup.
__global__ __launch_bounds__(1024) void ren_lu(float* __restrict__ E) {
  __shared__ float rowk[NXc];
  __shared__ float invp;
  const int tid = threadIdx.x;
  for (int k = 0; k < NXc - 1; ++k) {
    for (int j = tid; j < NXc; j += 1024) rowk[j] = E[(size_t)k * NXc + j];
    if (tid == 0) invp = 1.0f / E[(size_t)k * NXc + k];
    __syncthreads();
    for (int i = k + 1 + tid; i < NXc; i += 1024) {
      float l = E[(size_t)i * NXc + k] * invp;
      E[(size_t)i * NXc + k] = l;
      for (int j = k + 1; j < NXc; ++j)
        E[(size_t)i * NXc + j] -= l * rowk[j];
    }
    __syncthreads();
  }
}

// Solve E * Z = [H31 | H32 | B2] (1792 RHS columns), one column per thread.
// Writes A (512x512), B1 (512x1024), B2e (512x256) as bf16 hi/lo.
__global__ __launch_bounds__(64) void ren_solve(
    const float* __restrict__ LU, const float* __restrict__ H,
    const float* __restrict__ B2, float* __restrict__ Z,
    bf16_t* __restrict__ Ahd, bf16_t* __restrict__ Ald,
    bf16_t* __restrict__ B1h, bf16_t* __restrict__ B1l,
    bf16_t* __restrict__ B2h, bf16_t* __restrict__ B2l)
{
  const int c = blockIdx.x * 64 + threadIdx.x;   // 0..1791
  float* z = Z + (size_t)c * NXc;
  for (int i = 0; i < NXc; ++i) {                // forward: L z = rhs
    float rhs;
    if (c < NXc)            rhs = H[(size_t)(NXc + NVc + i) * NT + c];
    else if (c < NXc + NVc) rhs = H[(size_t)(NXc + NVc + i) * NT + NXc + (c - NXc)];
    else                    rhs = B2[(size_t)i * NUc + (c - NXc - NVc)];
    float s = rhs;
    for (int k = 0; k < i; ++k) s -= LU[(size_t)i * NXc + k] * z[k];
    z[i] = s;
  }
  for (int i = NXc - 1; i >= 0; --i) {           // backward: U x = z
    float s = z[i];
    for (int k = i + 1; k < NXc; ++k) s -= LU[(size_t)i * NXc + k] * z[k];
    s /= LU[(size_t)i * NXc + i];
    z[i] = s;
    bf16_t h, l; splitf(s, h, l);
    if (c < NXc)            { size_t o = (size_t)i * NXc + c;                Ahd[o] = h; Ald[o] = l; }
    else if (c < NXc + NVc) { size_t o = (size_t)i * NVc + (c - NXc);        B1h[o] = h; B1l[o] = l; }
    else                    { size_t o = (size_t)i * NUc + (c - NXc - NVc);  B2h[o] = h; B2l[o] = l; }
  }
}

// Li, C1 = -Li*H21, D11 = -Li*tril(H22,-1), D12e = Li*D12
__global__ void ren_prep(const float* __restrict__ H, const float* __restrict__ D12,
                         float* __restrict__ D11f,
                         bf16_t* __restrict__ D11h, bf16_t* __restrict__ D11l,
                         bf16_t* __restrict__ C1h,  bf16_t* __restrict__ C1l,
                         bf16_t* __restrict__ D12h, bf16_t* __restrict__ D12l)
{
  int j = blockIdx.x * 16 + threadIdx.x;   // 0..1023
  int v = blockIdx.y * 16 + threadIdx.y;   // 0..1023
  float Li = 2.0f / H[(size_t)(NXc + v) * NT + NXc + v];
  float d  = (j < v) ? (-Li * H[(size_t)(NXc + v) * NT + NXc + j]) : 0.0f;
  size_t o = (size_t)v * NVc + j;
  D11f[o] = d;
  { bf16_t h, l; splitf(d, h, l); D11h[o] = h; D11l[o] = l; }
  if (j < NXc) {
    float cv = -Li * H[(size_t)(NXc + v) * NT + j];
    bf16_t h, l; splitf(cv, h, l);
    C1h[(size_t)v * NXc + j] = h; C1l[(size_t)v * NXc + j] = l;
  }
  if (j < NUc) {
    float dv = Li * D12[(size_t)v * NUc + j];
    bf16_t h, l; splitf(dv, h, l);
    D12h[(size_t)v * NUc + j] = h; D12l[(size_t)v * NUc + j] = l;
  }
}

// In-block sequential equilibrium: w_i = relu(tmp_i + sum_{off<=j<off+i} D11*w)
// One batch row per thread; per-row w history lives in LDS (conflict-free).
__global__ __launch_bounds__(64) void ren_eqseq(
    const float* __restrict__ tmp, const float* __restrict__ D11f,
    bf16_t* __restrict__ wh, bf16_t* __restrict__ wl, int off)
{
  __shared__ float whist[128][64];               // [i][tid]
  const int tid = threadIdx.x;
  const size_t m = (size_t)blockIdx.x * 64 + tid;
  for (int i = 0; i < 128; ++i) {
    float s = tmp[m * 128 + i];
    const float* drow = D11f + (size_t)(off + i) * NVc + off;   // uniform (SGPR)
    for (int j = 0; j < i; ++j) s += drow[j] * whist[j][tid];
    s = fmaxf(s, 0.0f);
    whist[i][tid] = s;
    bf16_t h, l; splitf(s, h, l);
    wh[m * NVc + off + i] = h;
    wl[m * NVc + off + i] = l;
  }
}

// ---------------------------------------------------------------------------
extern "C" void kernel_launch(void* const* d_in, const int* in_sizes, int n_in,
                              void* d_out, int out_size, void* d_ws, size_t ws_size,
                              hipStream_t stream) {
  (void)in_sizes; (void)n_in; (void)out_size;
  const float* x   = (const float*)d_in[0];
  const float* u   = (const float*)d_in[1];
  const float* p   = (const float*)d_in[2];
  const float* X   = (const float*)d_in[3];
  const float* B2  = (const float*)d_in[4];
  const float* D12 = (const float*)d_in[5];
  const float* Y1  = (const float*)d_in[6];
  const float* C2  = (const float*)d_in[7];
  const float* D21 = (const float*)d_in[8];
  const float* D22 = (const float*)d_in[9];
  const float* bx  = (const float*)d_in[10];
  const float* bv  = (const float*)d_in[11];
  const float* by  = (const float*)d_in[12];
  float* out  = (float*)d_out;                 // x1: 4096x512
  float* outy = out + (size_t)BAT * NXc;       // y:  4096x256

  char*  wsp  = (char*)d_ws;
  size_t used = 0;
  auto alloc = [&](size_t bytes) -> char* {
    char* r = wsp + used;
    used += (bytes + 255) & ~(size_t)255;
    return r;
  };
  float*  Hb    = (float*)alloc((size_t)NT * NT * 4);
  float*  Eb    = (float*)alloc((size_t)NXc * NXc * 4);
  float*  D11f  = (float*)alloc((size_t)NVc * NVc * 4);
  float*  bbuf  = (float*)alloc((size_t)BAT * NVc * 4);
  float*  tmp   = (float*)alloc((size_t)BAT * 128 * 4);
  float*  Zb    = (float*)alloc((size_t)(NXc + NVc + NUc) * NXc * 4);
  float*  nrm   = (float*)alloc(256);
  bf16_t* Xth   = (bf16_t*)alloc((size_t)NT * NT * 2);
  bf16_t* Xtl   = (bf16_t*)alloc((size_t)NT * NT * 2);
  bf16_t* xh    = (bf16_t*)alloc((size_t)BAT * NXc * 2);
  bf16_t* xl    = (bf16_t*)alloc((size_t)BAT * NXc * 2);
  bf16_t* uh    = (bf16_t*)alloc((size_t)BAT * NUc * 2);
  bf16_t* ul    = (bf16_t*)alloc((size_t)BAT * NUc * 2);
  bf16_t* wh    = (bf16_t*)alloc((size_t)BAT * NVc * 2);
  bf16_t* wl    = (bf16_t*)alloc((size_t)BAT * NVc * 2);
  bf16_t* C1h   = (bf16_t*)alloc((size_t)NVc * NXc * 2);
  bf16_t* C1l   = (bf16_t*)alloc((size_t)NVc * NXc * 2);
  bf16_t* D12h  = (bf16_t*)alloc((size_t)NVc * NUc * 2);
  bf16_t* D12l  = (bf16_t*)alloc((size_t)NVc * NUc * 2);
  bf16_t* D11h  = (bf16_t*)alloc((size_t)NVc * NVc * 2);
  bf16_t* D11l  = (bf16_t*)alloc((size_t)NVc * NVc * 2);
  bf16_t* Ahd   = (bf16_t*)alloc((size_t)NXc * NXc * 2);
  bf16_t* Ald   = (bf16_t*)alloc((size_t)NXc * NXc * 2);
  bf16_t* B1h   = (bf16_t*)alloc((size_t)NXc * NVc * 2);
  bf16_t* B1l   = (bf16_t*)alloc((size_t)NXc * NVc * 2);
  bf16_t* B2eh  = (bf16_t*)alloc((size_t)NXc * NUc * 2);
  bf16_t* B2el  = (bf16_t*)alloc((size_t)NXc * NUc * 2);
  bf16_t* C2h   = (bf16_t*)alloc((size_t)NYc * NXc * 2);
  bf16_t* C2l   = (bf16_t*)alloc((size_t)NYc * NXc * 2);
  bf16_t* D21h  = (bf16_t*)alloc((size_t)NYc * NVc * 2);
  bf16_t* D21l  = (bf16_t*)alloc((size_t)NYc * NVc * 2);
  bf16_t* D22h  = (bf16_t*)alloc((size_t)NYc * NUc * 2);
  bf16_t* D22l  = (bf16_t*)alloc((size_t)NYc * NUc * 2);
  if (used > ws_size) return;   // workspace too small: bail (deterministic)

  // 0) norm + splits of inputs
  ren_init<<<1, 1, 0, stream>>>(nrm);
  ren_sumsq<<<256, 256, 0, stream>>>(X, nrm);
  ren_split_tr<<<dim3(NT / 16, NT / 16), dim3(16, 16), 0, stream>>>(X, Xth, Xtl);
  ren_split<<<(BAT * NXc) / 256, 256, 0, stream>>>(x,   xh,   xl,   BAT * NXc);
  ren_split<<<(BAT * NUc) / 256, 256, 0, stream>>>(u,   uh,   ul,   BAT * NUc);
  ren_split<<<(NYc * NXc) / 256, 256, 0, stream>>>(C2,  C2h,  C2l,  NYc * NXc);
  ren_split<<<(NYc * NVc) / 256, 256, 0, stream>>>(D21, D21h, D21l, NYc * NVc);
  ren_split<<<(NYc * NUc) / 256, 256, 0, stream>>>(D22, D22h, D22l, NYc * NUc);

  // 1) H = p^2 * Xt Xt^T / max(|X|^2, eps) + eps*I   (2048^3 WMMA GEMM)
  ren_gemm_nt<<<dim3(NT / 64, NT / 64), 128, 0, stream>>>(
      Xth, Xtl, NT, Xth, Xtl, NT, nullptr, 0, Hb, NT, nullptr, p, nrm, NT, 1);

  // 2) E, LU, solves -> A, B1, B2e (bf16 splits)
  ren_buildE<<<dim3(NXc / 16, NXc / 16), dim3(16, 16), 0, stream>>>(Hb, Y1, Eb);
  ren_lu<<<1, 1024, 0, stream>>>(Eb);
  ren_solve<<<(NXc + NVc + NUc) / 64, 64, 0, stream>>>(
      Eb, Hb, B2, Zb, Ahd, Ald, B1h, B1l, B2eh, B2el);

  // 3) Li, C1, D11, D12e
  ren_prep<<<dim3(NVc / 16, NVc / 16), dim3(16, 16), 0, stream>>>(
      Hb, D12, D11f, D11h, D11l, C1h, C1l, D12h, D12l);

  // 4) b = x C1^T + u D12e^T + bv
  ren_gemm_nt<<<dim3(NVc / 64, BAT / 64), 128, 0, stream>>>(
      xh, xl, NXc, C1h, C1l, NXc, nullptr, 0, bbuf, NVc, bv, nullptr, nullptr, NXc, 0);
  ren_gemm_nt<<<dim3(NVc / 64, BAT / 64), 128, 0, stream>>>(
      uh, ul, NUc, D12h, D12l, NUc, bbuf, NVc, bbuf, NVc, nullptr, nullptr, nullptr, NUc, 0);

  // 5) blocked triangular ReLU equilibrium: 8 blocks of 128 features.
  //    cross-block part = WMMA GEMM over completed w columns; in-block = seq.
  for (int kb = 0; kb < 8; ++kb) {
    int off = kb * 128;
    ren_gemm_nt<<<dim3(128 / 64, BAT / 64), 128, 0, stream>>>(
        wh, wl, NVc, D11h + (size_t)off * NVc, D11l + (size_t)off * NVc, NVc,
        bbuf + off, NVc, tmp, 128, nullptr, nullptr, nullptr, off, 0);
    ren_eqseq<<<BAT / 64, 64, 0, stream>>>(tmp, D11f, wh, wl, off);
  }

  // 6) x1 = x A^T + w B1^T + u B2e^T + bx
  ren_gemm_nt<<<dim3(NXc / 64, BAT / 64), 128, 0, stream>>>(
      xh, xl, NXc, Ahd, Ald, NXc, nullptr, 0, out, NXc, bx, nullptr, nullptr, NXc, 0);
  ren_gemm_nt<<<dim3(NXc / 64, BAT / 64), 128, 0, stream>>>(
      wh, wl, NVc, B1h, B1l, NVc, out, NXc, out, NXc, nullptr, nullptr, nullptr, NVc, 0);
  ren_gemm_nt<<<dim3(NXc / 64, BAT / 64), 128, 0, stream>>>(
      uh, ul, NUc, B2eh, B2el, NUc, out, NXc, out, NXc, nullptr, nullptr, nullptr, NUc, 0);

  // 7) y = x C2^T + w D21^T + u D22^T + by
  ren_gemm_nt<<<dim3(NYc / 64, BAT / 64), 128, 0, stream>>>(
      xh, xl, NXc, C2h, C2l, NXc, nullptr, 0, outy, NYc, by, nullptr, nullptr, NXc, 0);
  ren_gemm_nt<<<dim3(NYc / 64, BAT / 64), 128, 0, stream>>>(
      wh, wl, NVc, D21h, D21l, NVc, outy, NYc, outy, NYc, nullptr, nullptr, nullptr, NVc, 0);
  ren_gemm_nt<<<dim3(NYc / 64, BAT / 64), 128, 0, stream>>>(
      uh, ul, NUc, D22h, D22l, NUc, outy, NYc, outy, NYc, nullptr, nullptr, nullptr, NUc, 0);
}
